// Decoder_1400159338703
// MI455X (gfx1250) — matrix operand order repaired and checked by reading, not compiled, not run on hardware
//
#include <hip/hip_runtime.h>
#include <hip/hip_bf16.h>

typedef _Float16     __attribute__((ext_vector_type(16))) v16h;
typedef float        __attribute__((ext_vector_type(8)))  v8f;
typedef unsigned int __attribute__((ext_vector_type(8)))  v8u;
typedef unsigned int __attribute__((ext_vector_type(4)))  v4u;  // 16B-aligned LDS chunk
typedef unsigned int __attribute__((ext_vector_type(2)))  v2u;  // 8B-aligned LDS chunk

#define BATCH 512
#define SEQL  1024
#define UNITS 256
#define OUTD  32
#define BT    16          // batch rows per workgroup
#define NTHR  256         // 8 wave32 waves

// layer 0: x = [dt(1) | h0(116)] -> h0(116)
#define H0 116
#define C0 117
#define WS0 120           // stride multiple of 8 halves -> 16B-aligned ds_load_b128
#define XS0 128
#define NK0 4
// layer 1: x = [h0(116) | h1(76)] -> h1(76)
#define H1 76
#define C1 192
#define WS1 192
#define XS1 192
#define NK1 6
// layer 2: x = [h1(76) | h2(64)] -> h2(64)
#define H2 64
#define C2 140
#define WS2 144
#define XS2 160
#define NK2 5

#define SZW0 (H0*WS0 + 32)
#define SZW1 (H1*WS1 + 32)
#define SZW2 (H2*WS2 + 32)
#define SZWO (OUTD*64)
#define SZX0 (BT*XS0)
#define SZX1 (BT*XS1)
#define SZX2 (BT*XS2)
#define TOT_HALFS (4*SZW0 + 4*SZW1 + 4*SZW2 + SZWO + SZX0 + SZX1 + SZX2)
#define NBIAS (4*H0 + 4*H1 + 4*H2 + OUTD)
#define SMEM_BYTES (TOT_HALFS*2 + NBIAS*4)   // 326,272 B <= 320KB WGP LDS

// ---- WMMA fragment loaders (CDNA5 16x16x32 f16 layouts per ISA 7.12.2) ----

// A 16x32 f16: lane m = l&15, hi = l>>4; halves j<8 -> K=k0+hi*8+j,
//              j>=8 -> K=k0+16+hi*8+(j-8). Two 16B groups, 16B-aligned
//              when xs%8==0 and k0%8==0 (true for all main-path calls).
__device__ __forceinline__ v16h loadfrag_A16(const _Float16* X, int xs, int k0) {
  const int lane = threadIdx.x & 31;
  const int m = lane & 15, hi = lane >> 4;
  const v4u* p0 = (const v4u*)(X + m * xs + k0 + hi * 8);
  const v4u* p1 = (const v4u*)(X + m * xs + k0 + 16 + hi * 8);
  const v4u a = *p0, b = *p1;
  v8u u;
  u[0] = a[0]; u[1] = a[1]; u[2] = a[2]; u[3] = a[3];
  u[4] = b[0]; u[5] = b[1]; u[6] = b[2]; u[7] = b[3];
  return __builtin_bit_cast(v16h, u);
}

// Same layout, 8B-aligned variant (readout reads h2 at k0=76).
__device__ __forceinline__ v16h loadfrag_A8(const _Float16* X, int xs, int k0) {
  const int lane = threadIdx.x & 31;
  const int m = lane & 15, hi = lane >> 4;
  const v2u* p0 = (const v2u*)(X + m * xs + k0 + hi * 8);
  const v2u* p1 = (const v2u*)(X + m * xs + k0 + 16 + hi * 8);
  const v2u a0 = p0[0], a1 = p0[1], b0 = p1[0], b1 = p1[1];
  v8u u;
  u[0] = a0[0]; u[1] = a0[1]; u[2] = a1[0]; u[3] = a1[1];
  u[4] = b0[0]; u[5] = b0[1]; u[6] = b1[0]; u[7] = b1[1];
  return __builtin_bit_cast(v16h, u);
}

// B 32x16 f16 (B[k][n] = W[n][k]): lane n = n0+(l&15), hi = l>>4;
// halves j<8 -> K = k0 + hi*8 + j, j>=8 -> K = k0 + 16 + hi*8 + (j-8).
// Two aligned b128 loads (ws%8==0 guaranteed by padded strides).
__device__ __forceinline__ v16h loadfrag_B(const _Float16* W, int ws, int n0,
                                           int k0, int hmax) {
  const int lane = threadIdx.x & 31;
  int n = n0 + (lane & 15);
  if (n >= hmax) n = hmax - 1;      // stay in-bounds; garbage cols never used
  const int hi = lane >> 4;
  const v4u* p0 = (const v4u*)(W + n * ws + k0 + hi * 8);
  const v4u* p1 = (const v4u*)(W + n * ws + k0 + 16 + hi * 8);
  const v4u a = *p0, b = *p1;
  v8u u;
  u[0] = a[0]; u[1] = a[1]; u[2] = a[2]; u[3] = a[3];
  u[4] = b[0]; u[5] = b[1]; u[6] = b[2]; u[7] = b[3];
  return __builtin_bit_cast(v16h, u);
}

__device__ __forceinline__ v8f wmma_acc(v8f c, v16h a, v16h b) {
  return __builtin_amdgcn_wmma_f32_16x16x32_f16(false, a, false, b,
                                                (short)0, c, false, false);
}

// 4 matmuls of one CfC cell for this wave's N-tile
__device__ __forceinline__ void cell_mm(const _Float16* X, int xs, int nkt,
                                        const _Float16* w1, const _Float16* w2,
                                        const _Float16* wa, const _Float16* wb,
                                        int ws, int hmax, int n0,
                                        v8f& s1, v8f& s2, v8f& sa, v8f& sb) {
  for (int kt = 0; kt < nkt; ++kt) {
    const int k0 = kt * 32;
    v16h a = loadfrag_A16(X, xs, k0);
    v16h b;
    b = loadfrag_B(w1, ws, n0, k0, hmax); s1 = wmma_acc(s1, a, b);
    b = loadfrag_B(w2, ws, n0, k0, hmax); s2 = wmma_acc(s2, a, b);
    b = loadfrag_B(wa, ws, n0, k0, hmax); sa = wmma_acc(sa, a, b);
    b = loadfrag_B(wb, ws, n0, k0, hmax); sb = wmma_acc(sb, a, b);
  }
}

// CfC gate fusion + f16 state write-back (C/D layout: lane n, VGPR r = row).
// Biases pre-hoisted into registers by the caller.
__device__ __forceinline__ void cell_fuse(v8f s1, v8f s2, v8f sa, v8f sb,
                                          float c1, float c2, float ca, float cb,
                                          int n0, int hmax,
                                          _Float16* dA, int sA, int oA,
                                          _Float16* dB, int sB, int oB) {
  const int lane = threadIdx.x & 31;
  const int n = n0 + (lane & 15);
  const int hi = lane >> 4;
  if (n < hmax) {
#pragma unroll
    for (int r = 0; r < 8; ++r) {
      const int m = hi * 8 + r;
      const float z = (sa[r] + ca) + (sb[r] + cb);
      const float sig = 1.0f / (1.0f + __expf(-z));
      const float f1 = tanhf(s1[r] + c1);
      const float f2 = tanhf(s2[r] + c2);
      const _Float16 h = (_Float16)(f1 * (1.0f - sig) + sig * f2);
      dA[m * sA + oA + n] = h;
      if (dB) dB[m * sB + oB + n] = h;
    }
  }
}

__device__ __forceinline__ void fill_w(_Float16* dst, int ws, const float* src,
                                       const int* mask, int h, int c) {
  for (int i = threadIdx.x; i < h * c; i += NTHR) {
    const int n = i / c, k = i - n * c;
    float v = src[i];
    if (mask) v *= (float)mask[i];
    dst[n * ws + k] = (_Float16)v;
  }
}
__device__ __forceinline__ void fill_b(float* dst, const float* src, int h) {
  for (int i = threadIdx.x; i < h; i += NTHR) dst[i] = src[i];
}

extern "C" __global__ void __launch_bounds__(NTHR)
cfc_wired_kernel(const float* __restrict__ g_dt, const float* __restrict__ g_hx,
                 const float* w1_0, const float* w2_0, const float* wa_0,
                 const float* wb_0, const float* b1_0, const float* b2_0,
                 const float* ba_0, const float* bb_0, const int* m_0,
                 const float* w1_1, const float* w2_1, const float* wa_1,
                 const float* wb_1, const float* b1_1, const float* b2_1,
                 const float* ba_1, const float* bb_1, const int* m_1,
                 const float* w1_2, const float* w2_2, const float* wa_2,
                 const float* wb_2, const float* b1_2, const float* b2_2,
                 const float* ba_2, const float* bb_2, const int* m_2,
                 const float* g_wout, const float* g_bout,
                 float* __restrict__ g_out) {
  extern __shared__ char smem[];
  _Float16* hb = (_Float16*)smem;
  _Float16* p = hb;
  _Float16* lw0 = p; p += 4 * SZW0;
  _Float16* lw1 = p; p += 4 * SZW1;
  _Float16* lw2 = p; p += 4 * SZW2;
  _Float16* lwo = p; p += SZWO;
  _Float16* X0  = p; p += SZX0;
  _Float16* X1  = p; p += SZX1;
  _Float16* X2  = p; p += SZX2;
  float* fb  = (float*)(hb + TOT_HALFS);
  float* lb0 = fb;
  float* lb1 = lb0 + 4 * H0;
  float* lb2 = lb1 + 4 * H1;
  float* lbo = lb2 + 4 * H2;

  const int tid = threadIdx.x;
  const int w = tid >> 5;
  const int lane = tid & 31;
  const int b0 = blockIdx.x * BT;

  // zero everything (covers K/N padding of X buffers and weight pads)
  for (int i = tid; i < TOT_HALFS; i += NTHR) hb[i] = (_Float16)0.0f;
  __syncthreads();

  // masked weights -> f16 LDS (one-time; weights stay resident all 1024 steps)
  fill_w(lw0 + 0 * SZW0, WS0, w1_0, m_0, H0, C0);
  fill_w(lw0 + 1 * SZW0, WS0, w2_0, m_0, H0, C0);
  fill_w(lw0 + 2 * SZW0, WS0, wa_0, nullptr, H0, C0);
  fill_w(lw0 + 3 * SZW0, WS0, wb_0, nullptr, H0, C0);
  fill_w(lw1 + 0 * SZW1, WS1, w1_1, m_1, H1, C1);
  fill_w(lw1 + 1 * SZW1, WS1, w2_1, m_1, H1, C1);
  fill_w(lw1 + 2 * SZW1, WS1, wa_1, nullptr, H1, C1);
  fill_w(lw1 + 3 * SZW1, WS1, wb_1, nullptr, H1, C1);
  fill_w(lw2 + 0 * SZW2, WS2, w1_2, m_2, H2, C2);
  fill_w(lw2 + 1 * SZW2, WS2, w2_2, m_2, H2, C2);
  fill_w(lw2 + 2 * SZW2, WS2, wa_2, nullptr, H2, C2);
  fill_w(lw2 + 3 * SZW2, WS2, wb_2, nullptr, H2, C2);
  fill_w(lwo, 64, g_wout, nullptr, OUTD, 64);
  fill_b(lb0 + 0 * H0, b1_0, H0); fill_b(lb0 + 1 * H0, b2_0, H0);
  fill_b(lb0 + 2 * H0, ba_0, H0); fill_b(lb0 + 3 * H0, bb_0, H0);
  fill_b(lb1 + 0 * H1, b1_1, H1); fill_b(lb1 + 1 * H1, b2_1, H1);
  fill_b(lb1 + 2 * H1, ba_1, H1); fill_b(lb1 + 3 * H1, bb_1, H1);
  fill_b(lb2 + 0 * H2, b1_2, H2); fill_b(lb2 + 1 * H2, b2_2, H2);
  fill_b(lb2 + 2 * H2, ba_2, H2); fill_b(lb2 + 3 * H2, bb_2, H2);
  fill_b(lbo, g_bout, OUTD);

  // initial hidden state -> f16 A-operand buffers (hx = [h0|h1|h2])
  for (int i = tid; i < BT * H0; i += NTHR) {
    const int m = i / H0, k = i - m * H0;
    const _Float16 v = (_Float16)g_hx[(size_t)(b0 + m) * UNITS + k];
    X0[m * XS0 + 1 + k] = v;  X1[m * XS1 + k] = v;
  }
  for (int i = tid; i < BT * H1; i += NTHR) {
    const int m = i / H1, k = i - m * H1;
    const _Float16 v = (_Float16)g_hx[(size_t)(b0 + m) * UNITS + H0 + k];
    X1[m * XS1 + H0 + k] = v;  X2[m * XS2 + k] = v;
  }
  for (int i = tid; i < BT * H2; i += NTHR) {
    const int m = i / H2, k = i - m * H2;
    const _Float16 v = (_Float16)g_hx[(size_t)(b0 + m) * UNITS + H0 + H1 + k];
    X2[m * XS2 + H1 + k] = v;
  }
  if (tid < BT) X0[tid * XS0] = (_Float16)g_dt[(size_t)(b0 + tid) * SEQL];
  __syncthreads();

  // hoist per-lane biases into registers (constant across all 1024 steps)
  const int n0 = 16 * w;
  const int nn = n0 + (lane & 15);
  const int q0 = nn < H0 ? nn : H0 - 1;
  const int q1 = nn < H1 ? nn : H1 - 1;
  const int q2 = nn < H2 ? nn : H2 - 1;
  const int qo = nn < OUTD ? nn : OUTD - 1;
  const float B01 = lb0[q0], B02 = lb0[H0 + q0],
              B0a = lb0[2 * H0 + q0], B0b = lb0[3 * H0 + q0];
  const float B11 = lb1[q1], B12 = lb1[H1 + q1],
              B1a = lb1[2 * H1 + q1], B1b = lb1[3 * H1 + q1];
  const float B21 = lb2[q2], B22 = lb2[H2 + q2],
              B2a = lb2[2 * H2 + q2], B2b = lb2[3 * H2 + q2];
  const float BO = lbo[qo];

  for (int t = 0; t < SEQL; ++t) {
    // ---- layer 0: 8 N-tiles, one per wave ----
    {
      v8f s1 = {}, s2 = {}, sa = {}, sb = {};
      cell_mm(X0, XS0, NK0, lw0 + 0 * SZW0, lw0 + 1 * SZW0, lw0 + 2 * SZW0,
              lw0 + 3 * SZW0, WS0, H0, n0, s1, s2, sa, sb);
      __syncthreads();
      cell_fuse(s1, s2, sa, sb, B01, B02, B0a, B0b,
                n0, H0, X0, XS0, 1, X1, XS1, 0);
      __syncthreads();
    }
    // ---- layer 1: 5 N-tiles (waves 0..4) ----
    {
      v8f s1 = {}, s2 = {}, sa = {}, sb = {};
      if (w < 5)
        cell_mm(X1, XS1, NK1, lw1 + 0 * SZW1, lw1 + 1 * SZW1, lw1 + 2 * SZW1,
                lw1 + 3 * SZW1, WS1, H1, n0, s1, s2, sa, sb);
      __syncthreads();
      if (w < 5)
        cell_fuse(s1, s2, sa, sb, B11, B12, B1a, B1b,
                  n0, H1, X1, XS1, H0, X2, XS2, 0);
      __syncthreads();
    }
    // ---- layer 2: 4 N-tiles (waves 0..3); wave 4 prefetches next dt ----
    {
      v8f s1 = {}, s2 = {}, sa = {}, sb = {};
      if (w < 4)
        cell_mm(X2, XS2, NK2, lw2 + 0 * SZW2, lw2 + 1 * SZW2, lw2 + 2 * SZW2,
                lw2 + 3 * SZW2, WS2, H2, n0, s1, s2, sa, sb);
      __syncthreads();
      if (w < 4)
        cell_fuse(s1, s2, sa, sb, B21, B22, B2a, B2b,
                  n0, H2, X2, XS2, H1, nullptr, 0, 0);
      if (w == 4 && lane < BT && t + 1 < SEQL)
        X0[lane * XS0] = (_Float16)g_dt[(size_t)(b0 + lane) * SEQL + t + 1];
      __syncthreads();
    }
    // ---- fused readout: h2(16x64) @ Wout^T(64x32) + bout -> global ----
    if (w < 2) {
      v8f acc = {};
#pragma unroll
      for (int kt = 0; kt < 2; ++kt) {
        v16h a = loadfrag_A8(X2, XS2, H1 + 32 * kt);  // k0=76/108: 8B aligned
        v16h b = loadfrag_B(lwo, 64, n0, 32 * kt, OUTD);
        acc = wmma_acc(acc, a, b);
      }
      const int hi = lane >> 4;
#pragma unroll
      for (int r = 0; r < 8; ++r) {
        const int m = hi * 8 + r;
        g_out[((size_t)(b0 + m) * SEQL + t) * OUTD + nn] = acc[r] + BO;
      }
    }
  }
}

extern "C" void kernel_launch(void* const* d_in, const int* in_sizes, int n_in,
                              void* d_out, int out_size, void* d_ws,
                              size_t ws_size, hipStream_t stream) {
  (void)in_sizes; (void)n_in; (void)out_size; (void)d_ws; (void)ws_size;
  hipFuncSetAttribute((const void*)cfc_wired_kernel,
                      hipFuncAttributeMaxDynamicSharedMemorySize, SMEM_BYTES);
  cfc_wired_kernel<<<dim3(BATCH / BT), dim3(NTHR), SMEM_BYTES, stream>>>(
      (const float*)d_in[0],  (const float*)d_in[1],
      (const float*)d_in[2],  (const float*)d_in[3],  (const float*)d_in[4],
      (const float*)d_in[5],  (const float*)d_in[6],  (const float*)d_in[7],
      (const float*)d_in[8],  (const float*)d_in[9],  (const int*)d_in[10],
      (const float*)d_in[11], (const float*)d_in[12], (const float*)d_in[13],
      (const float*)d_in[14], (const float*)d_in[15], (const float*)d_in[16],
      (const float*)d_in[17], (const float*)d_in[18], (const int*)d_in[19],
      (const float*)d_in[20], (const float*)d_in[21], (const float*)d_in[22],
      (const float*)d_in[23], (const float*)d_in[24], (const float*)d_in[25],
      (const float*)d_in[26], (const float*)d_in[27], (const int*)d_in[28],
      (const float*)d_in[29], (const float*)d_in[30],
      (float*)d_out);
}
